// BertEmbedding_53085795778899
// MI455X (gfx1250) — matrix operand order repaired and checked by reading, not compiled, not run on hardware
//
#include <hip/hip_runtime.h>
#include <hip/hip_bf16.h>

// ---------------------------------------------------------------------------
// BERT-base encoder for MI455X (gfx1250, wave32, WMMA bf16 16x16x32).
// Block-cooperative GEMM: a 256-thread block (8 waves) computes a 128x64
// output tile. The shared 32x64 weight k-slab is staged to LDS once per
// k-step (coalesced global loads -> v_perm bf16 packing -> conflict-free
// ds reads), so each weight element is fetched once per 128 output rows.
// ---------------------------------------------------------------------------

#define H   768
#define LAY 12
#define NH  12
#define HD  64
#define FF  3072
#define BZ  8
#define SQ  256
#define TT  128
#define M_ROWS (BZ * SQ)       // 2048
#define H3  (3 * H)            // 2304

typedef __attribute__((ext_vector_type(16))) __bf16        v16bf;
typedef __attribute__((ext_vector_type(8)))  float         v8f;
typedef __attribute__((ext_vector_type(8)))  unsigned int  v8u;

// Pack two f32 -> packed bf16 pair {lo=bf16(a), hi=bf16(b)} with one v_perm.
static __device__ __forceinline__ unsigned int pack_bf16(float a, float b) {
    unsigned int ua = __builtin_bit_cast(unsigned int, a) + 0x8000u;
    unsigned int ub = __builtin_bit_cast(unsigned int, b) + 0x8000u;
    return __builtin_amdgcn_perm(ub, ua, 0x07060302u);
}

static __device__ __forceinline__ float gelu_exact(float x) {
    return 0.5f * x * (1.0f + erff(x * 0.70710678118654752f));
}

// A fragment: 16x32 (MxK), fp32 source, element (m,k) at A[m*lda + kbase + k].
// Per-lane K runs are contiguous -> 4x float4 loads.
static __device__ __forceinline__ v16bf load_a_frag(const float* __restrict__ A,
                                                    int lda, int kbase, int lane) {
    int g = lane >> 4, m = lane & 15;
    const float* base = A + (size_t)m * lda + kbase + g * 16;
    float4 f0 = *(const float4*)(base);          // k = g*16 + 0..3
    float4 f1 = *(const float4*)(base + 4);      // k = g*16 + 4..7
    float4 f2 = *(const float4*)(base + 8);      // k = g*16 + 8..11
    float4 f3 = *(const float4*)(base + 12);     // k = g*16 + 12..15
    v8u w;
    w[0] = pack_bf16(f0.x, f0.y);
    w[1] = pack_bf16(f0.z, f0.w);
    w[2] = pack_bf16(f1.x, f1.y);
    w[3] = pack_bf16(f1.z, f1.w);
    w[4] = pack_bf16(f2.x, f2.y);
    w[5] = pack_bf16(f2.z, f2.w);
    w[6] = pack_bf16(f3.x, f3.y);
    w[7] = pack_bf16(f3.z, f3.w);
    return __builtin_bit_cast(v16bf, w);
}

// B fragment from TRANSPOSED source: element (k,n) at Bt[n*ldb + kbase + k].
static __device__ __forceinline__ v16bf load_bt_frag(const float* __restrict__ Bt,
                                                     int ldb, int kbase, int lane) {
    int g = lane >> 4, n = lane & 15;
    const float* base = Bt + (size_t)n * ldb + kbase + g * 16;
    float4 f0 = *(const float4*)(base);
    float4 f1 = *(const float4*)(base + 4);
    float4 f2 = *(const float4*)(base + 8);
    float4 f3 = *(const float4*)(base + 12);
    v8u w;
    w[0] = pack_bf16(f0.x, f0.y);
    w[1] = pack_bf16(f0.z, f0.w);
    w[2] = pack_bf16(f1.x, f1.y);
    w[3] = pack_bf16(f1.z, f1.w);
    w[4] = pack_bf16(f2.x, f2.y);
    w[5] = pack_bf16(f2.z, f2.w);
    w[6] = pack_bf16(f3.x, f3.y);
    w[7] = pack_bf16(f3.z, f3.w);
    return __builtin_bit_cast(v16bf, w);
}

#define WMMA_BF16(a, b, c) \
    __builtin_amdgcn_wmma_f32_16x16x32_bf16(false, (a), false, (b), (short)0, (c), false, false)

// LDS slab: 16 packed-k2 rows x 64 cols, stride 66 u32 for bank-conflict-free
// fragment reads (lanes 0-15 -> banks c..c+15, lanes 16-31 -> c+16..c+31).
#define BSTRIDE 66

// Stage a 32x64 fp32 k-slab (rows kbase..kbase+31) of row-major B into LDS as
// packed bf16 pairs along K.  256 threads; 8 independent loads each, batched
// before packing so the compiler can overlap them.
static __device__ __forceinline__ void stage_b_slab(unsigned int* __restrict__ bslab,
                                                    const float* __restrict__ B,
                                                    int ldb, int kbase, int tid) {
    int n = tid & 63, k2b = tid >> 6;             // k2 = k2b + 4r, r=0..3
    float fa[4], fb[4];
#pragma unroll
    for (int r = 0; r < 4; ++r) {
        const float* gp = B + (size_t)(kbase + 2 * (k2b + 4 * r)) * ldb + n;
        fa[r] = gp[0];
        fb[r] = gp[ldb];
    }
#pragma unroll
    for (int r = 0; r < 4; ++r)
        bslab[(k2b + 4 * r) * BSTRIDE + n] = pack_bf16(fa[r], fb[r]);
}

// Read a 32x16 B fragment (column strip t) from the staged LDS slab.
static __device__ __forceinline__ v16bf load_b_frag_lds(const unsigned int* __restrict__ bslab,
                                                        int t, int lane) {
    int g = lane >> 4, n = lane & 15;
    const unsigned int* base = bslab + (size_t)(g * 8) * BSTRIDE + t * 16 + n;
    v8u w;
#pragma unroll
    for (int j = 0; j < 8; ++j) w[j] = base[j * BSTRIDE];
    return __builtin_bit_cast(v16bf, w);
}

// ---------------------------------------------------------------------------
// Block-cooperative WMMA GEMM: C[M,N] = act( A @ B + bias + residual ).
// Block = 8 waves = 128x64 output tile; wave w owns rows [w*16, w*16+16).
// ---------------------------------------------------------------------------
__global__ __launch_bounds__(256) void
wmma_gemm_kernel(const float* __restrict__ A, int lda,
                 const float* __restrict__ B, int ldb,
                 float* __restrict__ C, int ldc,
                 int M, int N, int K,
                 const float* __restrict__ bias,
                 const float* __restrict__ residual, int ldres,
                 int act) {
    __shared__ unsigned int bslab[16 * BSTRIDE];
    int tid  = threadIdx.x;
    int wave = tid >> 5;
    int lane = tid & 31;
    int nblk = N >> 6;
    int mb   = blockIdx.x / nblk;
    int cb   = blockIdx.x % nblk;

    const float* Ab = A + (size_t)(mb * 128 + wave * 16) * lda;
    const float* Bb = B + cb * 64;

    v8f acc[4];
#pragma unroll
    for (int t = 0; t < 4; ++t)
#pragma unroll
        for (int i = 0; i < 8; ++i) acc[t][i] = 0.0f;

    for (int k0 = 0; k0 < K; k0 += 32) {
        if (k0 + 32 < K)  // pull next weight k-slab toward L2/L0 while staging
            __builtin_prefetch(Bb + (size_t)(k0 + 32 + (tid >> 3)) * ldb + (tid & 7) * 8, 0, 1);
        stage_b_slab(bslab, Bb, ldb, k0, tid);
        __syncthreads();

        v16bf af = load_a_frag(Ab, lda, k0, lane);
#pragma unroll
        for (int t = 0; t < 4; ++t) {
            v16bf bf = load_b_frag_lds(bslab, t, lane);
            acc[t] = WMMA_BF16(af, bf, acc[t]);
        }
        __syncthreads();
    }

    int g = lane >> 4, n = lane & 15;
#pragma unroll
    for (int t = 0; t < 4; ++t) {
        int col = cb * 64 + t * 16 + n;
#pragma unroll
        for (int i = 0; i < 8; ++i) {
            int row = mb * 128 + wave * 16 + 8 * g + i;
            float v = acc[t][i];
            if (bias)     v += bias[col];
            if (residual) v += residual[(size_t)row * ldres + col];
            if (act == 1) v = gelu_exact(v);
            C[(size_t)row * ldc + col] = v;
        }
    }
}

// ---------------------------------------------------------------------------
// Attention scores: scores[b,h,q,k] = 0.125 * Q K^T.  Both operands load as
// contiguous float4 runs, so per-wave fragments straight from global are fine.
// One 16(q) x 64(k) strip per wave.
// ---------------------------------------------------------------------------
__global__ void attn_scores_kernel(const float* __restrict__ qkv,
                                   float* __restrict__ scores) {
    int wave = blockIdx.x * (blockDim.x >> 5) + (threadIdx.x >> 5);
    int lane = threadIdx.x & 31;
    const int ntiles = BZ * NH * 16 * 4;
    if (wave >= ntiles) return;                   // wave-uniform
    int bh = wave >> 6;
    int t  = wave & 63;
    int mt = t >> 2, cb = t & 3;
    int b = bh / NH, hh = bh % NH;

    const float* Aq = qkv + (size_t)(b * SQ + mt * 16) * H3 + hh * HD;

    v8f acc[4];
#pragma unroll
    for (int s = 0; s < 4; ++s)
#pragma unroll
        for (int i = 0; i < 8; ++i) acc[s][i] = 0.0f;

#pragma unroll
    for (int k0 = 0; k0 < HD; k0 += 32) {
        v16bf af = load_a_frag(Aq, H3, k0, lane);
#pragma unroll
        for (int s = 0; s < 4; ++s) {
            const float* Bk = qkv + (size_t)(b * SQ + cb * 64 + s * 16) * H3 + H + hh * HD;
            v16bf bf = load_bt_frag(Bk, H3, k0, lane);
            acc[s] = WMMA_BF16(af, bf, acc[s]);
        }
    }

    int g = lane >> 4, n = lane & 15;
    float* outp = scores + (size_t)bh * SQ * SQ;
#pragma unroll
    for (int s = 0; s < 4; ++s) {
#pragma unroll
        for (int i = 0; i < 8; ++i) {
            int q = mt * 16 + 8 * g + i;
            outp[(size_t)q * SQ + cb * 64 + s * 16 + n] = acc[s][i] * 0.125f;
        }
    }
}

// ---------------------------------------------------------------------------
// Masked softmax (rows = BZ*NH*SQ, cols = SQ = 256).
// ---------------------------------------------------------------------------
__global__ void softmax_kernel(float* __restrict__ scores,
                               const int* __restrict__ mask) {
    int row = blockIdx.x;
    int b   = row / (NH * SQ);
    int tid = threadIdx.x;
    __shared__ float red[256];

    float x = scores[(size_t)row * SQ + tid];
    x += (1.0f - (float)mask[b * SQ + tid]) * -10000.0f;

    red[tid] = x;
    __syncthreads();
    for (int st = 128; st > 0; st >>= 1) {
        if (tid < st) red[tid] = fmaxf(red[tid], red[tid + st]);
        __syncthreads();
    }
    float mx = red[0];
    __syncthreads();

    float e = expf(x - mx);
    red[tid] = e;
    __syncthreads();
    for (int st = 128; st > 0; st >>= 1) {
        if (tid < st) red[tid] += red[tid + st];
        __syncthreads();
    }
    float s = red[0];
    scores[(size_t)row * SQ + tid] = e / s;
}

// ---------------------------------------------------------------------------
// ctx = P @ V.  Block-cooperative: one block (8 waves) covers 128 q rows x
// the whole 64-wide head; the 32x64 V k-slab is LDS-staged and shared.
// Grid: BZ*NH*2 blocks (2 q-halves of 128 rows each).
// ---------------------------------------------------------------------------
__global__ __launch_bounds__(256) void
attn_av_kernel(const float* __restrict__ scores,
               const float* __restrict__ qkv,
               float* __restrict__ ctx) {
    __shared__ unsigned int bslab[16 * BSTRIDE];
    int tid  = threadIdx.x;
    int wave = tid >> 5;
    int lane = tid & 31;
    int bh   = blockIdx.x >> 1;
    int half = blockIdx.x & 1;
    int b = bh / NH, hh = bh % NH;
    int mt = half * 8 + wave;                      // q-tile of this wave

    const float* Ap = scores + (size_t)bh * SQ * SQ + (size_t)(mt * 16) * SQ;
    const float* Bv = qkv + (size_t)(b * SQ) * H3 + 2 * H + hh * HD;

    v8f acc[4];
#pragma unroll
    for (int t = 0; t < 4; ++t)
#pragma unroll
        for (int i = 0; i < 8; ++i) acc[t][i] = 0.0f;

    for (int k0 = 0; k0 < SQ; k0 += 32) {
        if (k0 + 32 < SQ)
            __builtin_prefetch(Bv + (size_t)(k0 + 32 + (tid >> 3)) * H3 + (tid & 7) * 8, 0, 1);
        stage_b_slab(bslab, Bv, H3, k0, tid);
        __syncthreads();

        v16bf af = load_a_frag(Ap, SQ, k0, lane);
#pragma unroll
        for (int t = 0; t < 4; ++t) {
            v16bf bf = load_b_frag_lds(bslab, t, lane);
            acc[t] = WMMA_BF16(af, bf, acc[t]);
        }
        __syncthreads();
    }

    int g = lane >> 4, n = lane & 15;
#pragma unroll
    for (int t = 0; t < 4; ++t) {
#pragma unroll
        for (int i = 0; i < 8; ++i) {
            int q = mt * 16 + 8 * g + i;
            ctx[(size_t)(b * SQ + q) * H + hh * HD + t * 16 + n] = acc[t][i];
        }
    }
}

// ---------------------------------------------------------------------------
// Embedding gather (raw sum; LN applied separately).
// ---------------------------------------------------------------------------
__global__ void embed_kernel(const int* __restrict__ ids,
                             const int* __restrict__ segs,
                             const float* __restrict__ we,
                             const float* __restrict__ pe,
                             const float* __restrict__ te,
                             float* __restrict__ out) {
    int row = blockIdx.x;
    int s   = row % SQ;
    int id  = ids[row];
    int sg  = segs[row];
    for (int c = threadIdx.x; c < H; c += blockDim.x) {
        out[(size_t)row * H + c] =
            we[(size_t)id * H + c] + pe[(size_t)s * H + c] + te[(size_t)sg * H + c];
    }
}

// ---------------------------------------------------------------------------
// LayerNorm over H=768 (one row per block, 256 threads x 3 elems).
// ---------------------------------------------------------------------------
__global__ void ln_kernel(const float* __restrict__ x,
                          const float* __restrict__ g,
                          const float* __restrict__ bp,
                          float* __restrict__ out) {
    int row = blockIdx.x;
    int tid = threadIdx.x;
    const float* xr = x + (size_t)row * H;
    float v0 = xr[tid], v1 = xr[tid + 256], v2 = xr[tid + 512];

    __shared__ float red[256];
    red[tid] = v0 + v1 + v2;
    __syncthreads();
    for (int st = 128; st > 0; st >>= 1) {
        if (tid < st) red[tid] += red[tid + st];
        __syncthreads();
    }
    float mean = red[0] * (1.0f / (float)H);
    __syncthreads();

    float d0 = v0 - mean, d1 = v1 - mean, d2 = v2 - mean;
    red[tid] = d0 * d0 + d1 * d1 + d2 * d2;
    __syncthreads();
    for (int st = 128; st > 0; st >>= 1) {
        if (tid < st) red[tid] += red[tid + st];
        __syncthreads();
    }
    float inv = rsqrtf(red[0] * (1.0f / (float)H) + 1e-12f);

    float* orow = out + (size_t)row * H;
    orow[tid]       = d0 * inv * g[tid]       + bp[tid];
    orow[tid + 256] = d1 * inv * g[tid + 256] + bp[tid + 256];
    orow[tid + 512] = d2 * inv * g[tid + 512] + bp[tid + 512];
}

// ---------------------------------------------------------------------------
// Ragged mean-pool.
// ---------------------------------------------------------------------------
__global__ void pool_kernel(const float* __restrict__ enc,
                            const int* __restrict__ lens,
                            float* __restrict__ out) {
    int bt = blockIdx.x;
    int b = bt / TT, t = bt % TT;
    __shared__ int sh_start, sh_len;
    if (threadIdx.x == 0) {
        int s0 = 0;
        for (int j = 0; j < t; ++j) s0 += lens[b * TT + j];
        sh_start = s0;
        sh_len   = lens[b * TT + t];
    }
    __syncthreads();
    int st = sh_start, ln = sh_len;
    for (int c = threadIdx.x; c < H; c += blockDim.x) {
        float acc = 0.0f;
        for (int j = 0; j < ln; ++j)
            acc += enc[(size_t)(b * SQ + st + j) * H + c];
        out[(size_t)bt * H + c] = (ln > 0) ? (acc / (float)ln) : 0.0f;
    }
}

// ---------------------------------------------------------------------------
// Host orchestration
// ---------------------------------------------------------------------------
static inline void launch_gemm(const float* A, int lda, const float* B, int ldb,
                               float* C, int ldc, int M, int N, int K,
                               const float* bias, const float* residual, int ldres,
                               int act, hipStream_t stream) {
    int blocks = (M / 128) * (N / 64);            // all dims are exact multiples
    wmma_gemm_kernel<<<blocks, 256, 0, stream>>>(A, lda, B, ldb, C, ldc, M, N, K,
                                                 bias, residual, ldres, act);
}

extern "C" void kernel_launch(void* const* d_in, const int* in_sizes, int n_in,
                              void* d_out, int out_size, void* d_ws, size_t ws_size,
                              hipStream_t stream) {
    (void)in_sizes; (void)n_in; (void)out_size; (void)ws_size;

    const int*   bert_ids   = (const int*)d_in[0];
    const int*   segments   = (const int*)d_in[1];
    const int*   bert_mask  = (const int*)d_in[2];
    const int*   bert_lens  = (const int*)d_in[3];
    const float* word_emb   = (const float*)d_in[4];
    const float* pos_emb    = (const float*)d_in[5];
    const float* type_emb   = (const float*)d_in[6];
    const float* emb_ln_g   = (const float*)d_in[7];
    const float* emb_ln_b   = (const float*)d_in[8];
    const float* qkv_w      = (const float*)d_in[9];
    const float* qkv_b      = (const float*)d_in[10];
    const float* attn_out_w = (const float*)d_in[11];
    const float* attn_out_b = (const float*)d_in[12];
    const float* ln1_g      = (const float*)d_in[13];
    const float* ln1_b      = (const float*)d_in[14];
    const float* ff1_w      = (const float*)d_in[15];
    const float* ff1_b      = (const float*)d_in[16];
    const float* ff2_w      = (const float*)d_in[17];
    const float* ff2_b      = (const float*)d_in[18];
    const float* ln2_g      = (const float*)d_in[19];
    const float* ln2_b      = (const float*)d_in[20];
    float* out = (float*)d_out;

    // Workspace layout (floats); total ~17.3M floats (~69 MB).
    float* ws = (float*)d_ws;
    size_t off = 0;
    float* h      = ws + off; off += (size_t)M_ROWS * H;
    float* h2     = ws + off; off += (size_t)M_ROWS * H;
    float* tmp    = ws + off; off += (size_t)M_ROWS * H;
    float* ctx    = ws + off; off += (size_t)M_ROWS * H;
    float* qkv    = ws + off; off += (size_t)M_ROWS * H3;
    float* sc_ff  = ws + off;                     // scores / FF1 union (same size)
    float* scores = sc_ff;
    float* ffbuf  = sc_ff;

    embed_kernel<<<M_ROWS, 256, 0, stream>>>(bert_ids, segments, word_emb,
                                             pos_emb, type_emb, tmp);
    ln_kernel<<<M_ROWS, 256, 0, stream>>>(tmp, emb_ln_g, emb_ln_b, h);

    for (int l = 0; l < LAY; ++l) {
        const float* Wqkv = qkv_w      + (size_t)l * H * H3;
        const float* Bqkv = qkv_b      + (size_t)l * H3;
        const float* Wao  = attn_out_w + (size_t)l * H * H;
        const float* Bao  = attn_out_b + (size_t)l * H;
        const float* G1   = ln1_g      + (size_t)l * H;
        const float* B1   = ln1_b      + (size_t)l * H;
        const float* W1   = ff1_w      + (size_t)l * H * FF;
        const float* Bf1  = ff1_b      + (size_t)l * FF;
        const float* W2   = ff2_w      + (size_t)l * FF * H;
        const float* Bf2  = ff2_b      + (size_t)l * H;
        const float* G2   = ln2_g      + (size_t)l * H;
        const float* B2   = ln2_b      + (size_t)l * H;

        launch_gemm(h, H, Wqkv, H3, qkv, H3, M_ROWS, H3, H,
                    Bqkv, nullptr, 0, 0, stream);

        attn_scores_kernel<<<(BZ * NH * 64 + 7) / 8, 256, 0, stream>>>(qkv, scores);
        softmax_kernel<<<BZ * NH * SQ, 256, 0, stream>>>(scores, bert_mask);
        attn_av_kernel<<<BZ * NH * 2, 256, 0, stream>>>(scores, qkv, ctx);

        launch_gemm(ctx, H, Wao, H, tmp, H, M_ROWS, H, H,
                    Bao, h, H, 0, stream);
        ln_kernel<<<M_ROWS, 256, 0, stream>>>(tmp, G1, B1, h2);

        launch_gemm(h2, H, W1, FF, ffbuf, FF, M_ROWS, FF, H,
                    Bf1, nullptr, 0, 1, stream);

        launch_gemm(ffbuf, FF, W2, H, tmp, H, M_ROWS, H, FF,
                    Bf2, h2, H, 0, stream);
        ln_kernel<<<M_ROWS, 256, 0, stream>>>(tmp, G2, B2, h);
    }

    pool_kernel<<<BZ * TT, 256, 0, stream>>>(h, bert_lens, out);
}
